// LigRecConv_53309134077971
// MI455X (gfx1250) — matrix operand order repaired and checked by reading, not compiled, not run on hardware
//
#include <hip/hip_runtime.h>
#include <stdint.h>

// ---------------------------------------------------------------------------
// LigRecConv (EGNN-style edge conv) for MI455X / gfx1250.
// Compute-bound edge MLPs -> bf16 WMMA (16x16x32) with f32 accumulation.
// 64 edges per block (4 M-tiles) so each B-fragment feeds 4 WMMAs (L2 traffic
// /4 vs naive). Register use capped (waves_per_eu) to stay under 256 VGPRs.
// ---------------------------------------------------------------------------

typedef __attribute__((ext_vector_type(16))) __bf16 v16bf;
typedef __attribute__((ext_vector_type(8)))  float  v8f;

union FragU { uint4 q[2]; v16bf v; };

__device__ __forceinline__ unsigned short f2bf(float f) {
  unsigned int u = __float_as_uint(f);
  u += 0x7FFFu + ((u >> 16) & 1u);        // round-to-nearest-even
  return (unsigned short)(u >> 16);
}
__device__ __forceinline__ float sigm(float x) { return 1.0f / (1.0f + __expf(-x)); }
__device__ __forceinline__ float siluf(float x) { return x * sigm(x); }

// ---- pack W[K][ldN] (f32, row-major) into bf16 B-fragments ------------------
// frag = ks*NT + nt; per lane (n = lane&15, khalf = lane>>4): K = 8*khalf+e (e<8),
// 16+8*khalf+(e-8) (e>=8) -> 32B per lane (matches ISA bf16 B 32x16 layout).
__global__ void pack_b_frags(const float* __restrict__ W, int K, int ldN, int NT,
                             unsigned short* __restrict__ out) {
  int total = K * NT * 16;
  for (int idx = blockIdx.x * blockDim.x + threadIdx.x; idx < total;
       idx += gridDim.x * blockDim.x) {
    int elem = idx & 15;
    int lane = (idx >> 4) & 31;
    int frag = idx >> 9;
    int nt = frag % NT;
    int ks = frag / NT;
    int k = ks * 32 + (lane >> 4) * 8 + (elem & 7) + (elem >> 3) * 16;
    int n = nt * 16 + (lane & 15);
    float v = (n < ldN) ? W[k * ldN + n] : 0.0f;
    out[idx] = f2bf(v);
  }
}

__global__ void zero_f32(float* __restrict__ p, int n) {
  for (int i = blockIdx.x * blockDim.x + threadIdx.x; i < n;
       i += gridDim.x * blockDim.x)
    p[i] = 0.0f;
}

// ---- 2-layer MLP over 64 edges (4 M-tiles); each wave owns 2 N-tiles. -------
// fS: 64x256 bf16 (row stride 264). mS: 64x128 bf16 (row stride 136).
// On return: out2 = silu(layer2) f32 in regs AND bf16 copy in mS.
__device__ __forceinline__ void mlp2_wide(
    const unsigned short* fS, unsigned short* mS, const uint4* wq, int b1q,
    int b2q, const float* __restrict__ bias1, const float* __restrict__ bias2,
    const float* __restrict__ wlast, const float* dijS,
    int lane, int n, int hi, int wv, v8f out2[4][2]) {
  const int rowA = lane & 15;
  {
    v8f acc[4][2];
#pragma unroll
    for (int j = 0; j < 2; ++j) {
      float b0 = bias1[(2 * wv + j) * 16 + n];
#pragma unroll
      for (int tt = 0; tt < 4; ++tt)
#pragma unroll
        for (int r = 0; r < 8; ++r) acc[tt][j][r] = b0;
    }
#pragma unroll 2
    for (int ks = 0; ks < 8; ++ks) {        // K = 256
      FragU a[4], b[2];
      int cb = ks * 32 + hi * 8;
#pragma unroll
      for (int tt = 0; tt < 4; ++tt) {
        int ro = (tt * 16 + rowA) * 264 + cb;
        a[tt].q[0] = *(const uint4*)&fS[ro];
        a[tt].q[1] = *(const uint4*)&fS[ro + 16];
      }
#pragma unroll
      for (int j = 0; j < 2; ++j) {
        int fq = b1q + (ks * 8 + 2 * wv + j) * 64 + lane * 2;
        b[j].q[0] = wq[fq]; b[j].q[1] = wq[fq + 1];
      }
#pragma unroll
      for (int tt = 0; tt < 4; ++tt)
#pragma unroll
        for (int j = 0; j < 2; ++j)
          acc[tt][j] = __builtin_amdgcn_wmma_f32_16x16x32_bf16(
              false, a[tt].v, false, b[j].v, (short)0, acc[tt][j], false, false);
    }
    // rank-1 dij column + bias already in, SiLU, spill bf16 to mS
#pragma unroll
    for (int j = 0; j < 2; ++j) {
      int nt = 2 * wv + j;
      float wl = wlast[nt * 16 + n];
#pragma unroll
      for (int tt = 0; tt < 4; ++tt)
#pragma unroll
        for (int r = 0; r < 8; ++r) {
          int m = tt * 16 + r + 8 * hi;
          float x = acc[tt][j][r] + dijS[m] * wl;
          mS[m * 136 + nt * 16 + n] = f2bf(siluf(x));
        }
    }
  }
  __syncthreads();
  // layer 2 (accumulate straight into out2)
#pragma unroll
  for (int j = 0; j < 2; ++j) {
    float b0 = bias2[(2 * wv + j) * 16 + n];
#pragma unroll
    for (int tt = 0; tt < 4; ++tt)
#pragma unroll
      for (int r = 0; r < 8; ++r) out2[tt][j][r] = b0;
  }
#pragma unroll 2
  for (int ks = 0; ks < 4; ++ks) {          // K = 128
    FragU a[4], b[2];
    int cb = ks * 32 + hi * 8;
#pragma unroll
    for (int tt = 0; tt < 4; ++tt) {
      int ro = (tt * 16 + rowA) * 136 + cb;
      a[tt].q[0] = *(const uint4*)&mS[ro];
      a[tt].q[1] = *(const uint4*)&mS[ro + 16];
    }
#pragma unroll
    for (int j = 0; j < 2; ++j) {
      int fq = b2q + (ks * 8 + 2 * wv + j) * 64 + lane * 2;
      b[j].q[0] = wq[fq]; b[j].q[1] = wq[fq + 1];
    }
#pragma unroll
    for (int tt = 0; tt < 4; ++tt)
#pragma unroll
      for (int j = 0; j < 2; ++j)
        out2[tt][j] = __builtin_amdgcn_wmma_f32_16x16x32_bf16(
            false, a[tt].v, false, b[j].v, (short)0, out2[tt][j], false, false);
  }
  __syncthreads();                          // everyone done reading layer-1 mS
#pragma unroll
  for (int j = 0; j < 2; ++j) {
    int nt = 2 * wv + j;
#pragma unroll
    for (int tt = 0; tt < 4; ++tt)
#pragma unroll
      for (int r = 0; r < 8; ++r) {
        int m = tt * 16 + r + 8 * hi;
        float v = siluf(out2[tt][j][r]);
        out2[tt][j][r] = v;
        mS[m * 136 + nt * 16 + n] = f2bf(v);
      }
  }
  __syncthreads();
}

// ---- 128 -> 1 projection via WMMA (zero-padded 128x16 B); one M-tile/wave ---
__device__ __forceinline__ void col_gemm_t(const unsigned short* mS,
                                           const uint4* wq, int bq, float bias,
                                           int lane, int hi, int tt,
                                           float* outS, bool do_sig) {
  const int rowA = tt * 16 + (lane & 15);
  v8f acc;
#pragma unroll
  for (int r = 0; r < 8; ++r) acc[r] = bias;
#pragma unroll 2
  for (int ks = 0; ks < 4; ++ks) {
    FragU a, b;
    int cb = ks * 32 + hi * 8;
    a.q[0] = *(const uint4*)&mS[rowA * 136 + cb];
    a.q[1] = *(const uint4*)&mS[rowA * 136 + cb + 16];
    int fq = bq + ks * 64 + lane * 2;       // NT=1 -> frag index = ks
    b.q[0] = wq[fq]; b.q[1] = wq[fq + 1];
    acc = __builtin_amdgcn_wmma_f32_16x16x32_bf16(false, a.v, false, b.v,
                                                  (short)0, acc, false, false);
  }
  if ((lane & 15) == 0) {                   // column 0 lives in lanes 0 and 16
#pragma unroll
    for (int r = 0; r < 8; ++r) {
      float v = acc[r];
      outS[tt * 16 + r + 8 * hi] = do_sig ? sigm(v) : v;
    }
  }
}

// ---- edge kernel: 64 edges per block iteration ------------------------------
__global__ void __launch_bounds__(128)
__attribute__((amdgpu_waves_per_eu(4)))
edge_kernel(const float* __restrict__ hsrc, const float* __restrict__ hdst,
            const float* __restrict__ xsrc, const float* __restrict__ xdst,
            const int* __restrict__ srcI, const int* __restrict__ dstI,
            int nIters, const unsigned short* __restrict__ wf,
            const float* __restrict__ eb1, const float* __restrict__ eb2,
            const float* __restrict__ abp,
            const float* __restrict__ cb1, const float* __restrict__ cb2,
            const float* __restrict__ w1last, const float* __restrict__ c1last,
            float* __restrict__ hacc, float* __restrict__ xacc) {
  __shared__ unsigned short fS[64 * 264];   // 64 x 256 bf16 (+pad)  ~33.8KB
  __shared__ unsigned short mS[64 * 136];   // 64 x 128 bf16 (+pad)  ~17.4KB
  __shared__ int srcS[64], dstS[64];
  __shared__ float dijS[64], gateS[64], csS[64];
  __shared__ float xdS[64][3];

  const int t = threadIdx.x;
  const int lane = t & 31;
  const int wv = t >> 5;
  const int n = lane & 15;
  const int hi = lane >> 4;
  const uint4* wq = (const uint4*)wf;
  const float abv = abp[0];

  for (int blk = blockIdx.x; blk < nIters; blk += gridDim.x) {
    if (t < 64) {
      int e = blk * 64 + t;
      int s = srcI[e], d = dstI[e];
      srcS[t] = s; dstS[t] = d;
      float dx = xsrc[s * 3 + 0] - xdst[d * 3 + 0];
      float dy = xsrc[s * 3 + 1] - xdst[d * 3 + 1];
      float dz = xsrc[s * 3 + 2] - xdst[d * 3 + 2];
      float dij = sqrtf(dx * dx + dy * dy + dz * dz);
      float inv = 1.0f / (dij + 1.0f);
      dijS[t] = dij;
      xdS[t][0] = dx * inv; xdS[t][1] = dy * inv; xdS[t][2] = dz * inv;
    }
    __syncthreads();
    // gather f = [h_src | h_dst] -> bf16 LDS (64 edges x 64 float4 chunks)
#pragma unroll 4
    for (int it = 0; it < 32; ++it) {
      int chunk = it * 128 + t;
      int m = chunk >> 6;
      int p = chunk & 63;
      int q = p & 31;
      float4 v = (p < 32) ? ((const float4*)(hsrc + (size_t)srcS[m] * 128))[q]
                          : ((const float4*)(hdst + (size_t)dstS[m] * 128))[q];
      int col = ((p & 32) ? 128 : 0) + q * 4;
      unsigned short* o = &fS[m * 264 + col];
      o[0] = f2bf(v.x); o[1] = f2bf(v.y); o[2] = f2bf(v.z); o[3] = f2bf(v.w);
    }
    __syncthreads();

    // -------- h-message path: m2 = silu(silu(f@W1+b1)@W2+b2) ----------------
    v8f m2[4][2];
    mlp2_wide(fS, mS, wq, 0, 4096, eb1, eb2, w1last, dijS, lane, n, hi, wv, m2);
    col_gemm_t(mS, wq, 6144, abv, lane, hi, wv, gateS, true);   // aW gate
    __syncthreads();
#pragma unroll
    for (int tt = 0; tt < 4; ++tt)
#pragma unroll
      for (int j = 0; j < 2; ++j) {
        int nt = 2 * wv + j;
#pragma unroll
        for (int r = 0; r < 8; ++r) {
          int m = tt * 16 + r + 8 * hi;
          atomicAdd(&hacc[(size_t)dstS[m] * 128 + nt * 16 + n],
                    m2[tt][j][r] * gateS[m]);
        }
      }

    // -------- coord path: s = (silu(silu(f@cW1)@cW2)) @ cW3 -----------------
    v8f c2[4][2];
    mlp2_wide(fS, mS, wq, 6400, 10496, cb1, cb2, c1last, dijS, lane, n, hi, wv, c2);
    (void)c2;
    col_gemm_t(mS, wq, 12544, 0.0f, lane, hi, wv, csS, false);  // cW3
    __syncthreads();
    if (t < 64) {
      float s = csS[t];
      int d = dstS[t];
      atomicAdd(&xacc[d * 4 + 0], s * xdS[t][0]);
      atomicAdd(&xacc[d * 4 + 1], s * xdS[t][1]);
      atomicAdd(&xacc[d * 4 + 2], s * xdS[t][2]);
    }
    __syncthreads();
  }
}

// ---- node update kernel (16 nodes / block iter; ~1% of total FLOPs) ---------
__global__ void __launch_bounds__(128)
__attribute__((amdgpu_waves_per_eu(4)))
node_kernel(const float* __restrict__ h_lig, const float* __restrict__ x_lig,
            const float* __restrict__ z, const float* __restrict__ hacc,
            const float* __restrict__ xacc, const unsigned short* __restrict__ wf,
            const float* __restrict__ b1, const float* __restrict__ b2,
            float* __restrict__ h_out, float* __restrict__ x_out, int nTiles) {
  __shared__ unsigned short fS[16 * 264];
  __shared__ unsigned short mS[16 * 136];
  __shared__ float invzS[16];

  const int t = threadIdx.x;
  const int lane = t & 31;
  const int wv = t >> 5;
  const int n = lane & 15;
  const int hi = lane >> 4;
  const int aRowF = (lane & 15) * 264;
  const int aRowM = (lane & 15) * 136;
  const uint4* wq = (const uint4*)wf;

  for (int tile = blockIdx.x; tile < nTiles; tile += gridDim.x) {
    if (t < 16) invzS[t] = 1.0f / z[tile * 16 + t];
    __syncthreads();
#pragma unroll 2
    for (int it = 0; it < 8; ++it) {
      int chunk = it * 128 + t;
      int m = chunk >> 6;
      int p = chunk & 63;
      int q = p & 31;
      size_t node = (size_t)(tile * 16 + m);
      float4 v;
      int col;
      if (p < 32) { v = ((const float4*)(h_lig + node * 128))[q]; col = q * 4; }
      else {
        v = ((const float4*)(hacc + node * 128))[q];
        float s = invzS[m];
        v.x *= s; v.y *= s; v.z *= s; v.w *= s;
        col = 128 + q * 4;
      }
      unsigned short* o = &fS[m * 264 + col];
      o[0] = f2bf(v.x); o[1] = f2bf(v.y); o[2] = f2bf(v.z); o[3] = f2bf(v.w);
    }
    __syncthreads();
#pragma unroll
    for (int j = 0; j < 2; ++j) {           // layer 1: 256 -> 128
      int nt = 2 * wv + j;
      float b0 = b1[nt * 16 + n];
      v8f acc;
#pragma unroll
      for (int r = 0; r < 8; ++r) acc[r] = b0;
#pragma unroll 2
      for (int ks = 0; ks < 8; ++ks) {
        FragU a, b;
        int cb = ks * 32 + hi * 8;
        a.q[0] = *(const uint4*)&fS[aRowF + cb];
        a.q[1] = *(const uint4*)&fS[aRowF + cb + 16];
        int fq = (ks * 8 + nt) * 64 + lane * 2;
        b.q[0] = wq[fq]; b.q[1] = wq[fq + 1];
        acc = __builtin_amdgcn_wmma_f32_16x16x32_bf16(false, a.v, false, b.v,
                                                      (short)0, acc, false, false);
      }
#pragma unroll
      for (int r = 0; r < 8; ++r)
        mS[(r + 8 * hi) * 136 + nt * 16 + n] = f2bf(siluf(acc[r]));
    }
    __syncthreads();
#pragma unroll
    for (int j = 0; j < 2; ++j) {           // layer 2: 128 -> 128 + residual
      int nt = 2 * wv + j;
      float b0 = b2[nt * 16 + n];
      v8f acc;
#pragma unroll
      for (int r = 0; r < 8; ++r) acc[r] = b0;
#pragma unroll 2
      for (int ks = 0; ks < 4; ++ks) {
        FragU a, b;
        int cb = ks * 32 + hi * 8;
        a.q[0] = *(const uint4*)&mS[aRowM + cb];
        a.q[1] = *(const uint4*)&mS[aRowM + cb + 16];
        int fq = 4096 + (ks * 8 + nt) * 64 + lane * 2;
        b.q[0] = wq[fq]; b.q[1] = wq[fq + 1];
        acc = __builtin_amdgcn_wmma_f32_16x16x32_bf16(false, a.v, false, b.v,
                                                      (short)0, acc, false, false);
      }
#pragma unroll
      for (int r = 0; r < 8; ++r) {
        size_t node = (size_t)(tile * 16 + r + 8 * hi);
        int col = nt * 16 + n;
        h_out[node * 128 + col] = h_lig[node * 128 + col] + acc[r];
      }
    }
    if (t < 16) {
      size_t node = (size_t)(tile * 16 + t);
      float s = invzS[t];
#pragma unroll
      for (int c = 0; c < 3; ++c)
        x_out[node * 3 + c] = x_lig[node * 3 + c] + xacc[node * 4 + c] * s;
    }
    __syncthreads();
  }
}

extern "C" void kernel_launch(void* const* d_in, const int* in_sizes, int n_in,
                              void* d_out, int out_size, void* d_ws, size_t ws_size,
                              hipStream_t stream) {
  (void)in_sizes; (void)n_in; (void)out_size; (void)ws_size;
  const float* h_lig = (const float*)d_in[0];
  const float* h_kp  = (const float*)d_in[1];
  const float* x_lig = (const float*)d_in[2];
  const float* x_kp  = (const float*)d_in[3];
  const float* z_lig = (const float*)d_in[4];
  const float* ll_eW1 = (const float*)d_in[5];  const float* ll_eb1 = (const float*)d_in[6];
  const float* ll_eW2 = (const float*)d_in[7];  const float* ll_eb2 = (const float*)d_in[8];
  const float* ll_aW  = (const float*)d_in[9];  const float* ll_ab  = (const float*)d_in[10];
  const float* ll_cW1 = (const float*)d_in[11]; const float* ll_cb1 = (const float*)d_in[12];
  const float* ll_cW2 = (const float*)d_in[13]; const float* ll_cb2 = (const float*)d_in[14];
  const float* ll_cW3 = (const float*)d_in[15];
  const float* kl_eW1 = (const float*)d_in[16]; const float* kl_eb1 = (const float*)d_in[17];
  const float* kl_eW2 = (const float*)d_in[18]; const float* kl_eb2 = (const float*)d_in[19];
  const float* kl_aW  = (const float*)d_in[20]; const float* kl_ab  = (const float*)d_in[21];
  const float* kl_cW1 = (const float*)d_in[22]; const float* kl_cb1 = (const float*)d_in[23];
  const float* kl_cW2 = (const float*)d_in[24]; const float* kl_cb2 = (const float*)d_in[25];
  const float* kl_cW3 = (const float*)d_in[26];
  const float* n_W1 = (const float*)d_in[27]; const float* n_b1 = (const float*)d_in[28];
  const float* n_W2 = (const float*)d_in[29]; const float* n_b2 = (const float*)d_in[30];
  const int* ll_src = (const int*)d_in[31]; const int* ll_dst = (const int*)d_in[32];
  const int* kl_src = (const int*)d_in[33]; const int* kl_dst = (const int*)d_in[34];

  // workspace layout
  char* ws = (char*)d_ws;
  float* hacc = (float*)ws;                                   // 10000*128 f32
  float* xacc = (float*)(ws + (size_t)10000 * 128 * 4);       // 10000*4  f32
  unsigned short* wf_ll = (unsigned short*)(ws + (size_t)10000 * 128 * 4 + 10000 * 4 * 4);
  unsigned short* wf_kl = wf_ll + 102400;
  unsigned short* wf_nd = wf_kl + 102400;

  int zn = 10000 * 128 + 10000 * 4;
  zero_f32<<<(zn + 255) / 256, 256, 0, stream>>>(hacc, zn);

  auto pk = [&](const float* W, int K, int ldN, int NT, unsigned short* out) {
    int total = K * NT * 16;
    pack_b_frags<<<(total + 255) / 256, 256, 0, stream>>>(W, K, ldN, NT, out);
  };
  // bf16 offsets within each edge-type pack: eW1=0, eW2=32768, aW=49152,
  // cW1=51200, cW2=83968, cW3=100352  (uint4 offsets: 0,4096,6144,6400,10496,12544)
  pk(ll_eW1, 256, 128, 8, wf_ll + 0);
  pk(ll_eW2, 128, 128, 8, wf_ll + 32768);
  pk(ll_aW,  128, 1,   1, wf_ll + 49152);
  pk(ll_cW1, 256, 128, 8, wf_ll + 51200);
  pk(ll_cW2, 128, 128, 8, wf_ll + 83968);
  pk(ll_cW3, 128, 1,   1, wf_ll + 100352);
  pk(kl_eW1, 256, 128, 8, wf_kl + 0);
  pk(kl_eW2, 128, 128, 8, wf_kl + 32768);
  pk(kl_aW,  128, 1,   1, wf_kl + 49152);
  pk(kl_cW1, 256, 128, 8, wf_kl + 51200);
  pk(kl_cW2, 128, 128, 8, wf_kl + 83968);
  pk(kl_cW3, 128, 1,   1, wf_kl + 100352);
  pk(n_W1, 256, 128, 8, wf_nd + 0);
  pk(n_W2, 128, 128, 8, wf_nd + 32768);

  // 320000/64 = 5000 iters, 160000/64 = 2500 iters (exactly divisible)
  edge_kernel<<<5000, 128, 0, stream>>>(h_lig, h_lig, x_lig, x_lig, ll_src, ll_dst,
      5000, wf_ll, ll_eb1, ll_eb2, ll_ab, ll_cb1, ll_cb2,
      ll_eW1 + 256 * 128, ll_cW1 + 256 * 128, hacc, xacc);
  edge_kernel<<<2500, 128, 0, stream>>>(h_kp, h_lig, x_kp, x_lig, kl_src, kl_dst,
      2500, wf_kl, kl_eb1, kl_eb2, kl_ab, kl_cb1, kl_cb2,
      kl_eW1 + 256 * 128, kl_cW1 + 256 * 128, hacc, xacc);

  float* h_out = (float*)d_out;
  float* x_out = h_out + (size_t)10000 * 128;
  node_kernel<<<625, 128, 0, stream>>>(h_lig, x_lig, z_lig, hacc, xacc, wf_nd,
                                       n_b1, n_b2, h_out, x_out, 625);
}
